// FasterMultiHeadAttention_64003602645285
// MI455X (gfx1250) — matrix operand order repaired and checked by reading, not compiled, not run on hardware
//
#include <hip/hip_runtime.h>
#include <hip/hip_bf16.h>

#define BB 2
#define SS 2048
#define DD 1024
#define HH 16
#define HD 64
#define N3D (3*DD)
#define BS (BB*SS)

typedef __attribute__((ext_vector_type(16))) _Float16 v16h;
typedef __attribute__((ext_vector_type(8)))  float    v8f;
typedef __attribute__((ext_vector_type(8)))  _Float16 h8;

union V16H { v16h v; h8 h[2]; };

// ---- WMMA wrapper: D = A(16x32 f16) * B(32x16 f16) + C(16x16 f32) ----
__device__ __forceinline__ v8f wmma_f16(v16h a, v16h b, v8f c) {
  return __builtin_amdgcn_wmma_f32_16x16x32_f16(
      /*neg_a=*/false, a, /*neg_b=*/false, b,
      /*c_mod=*/(short)0, c, /*reuse_a=*/false, /*reuse_b=*/false);
}

// ---- CDNA5 async global->LDS copy (ASYNCcnt) ----
__device__ __forceinline__ void async_ld_b128(unsigned lds_off, const void* g) {
  asm volatile("global_load_async_to_lds_b128 %0, %1, off"
               :: "v"(lds_off), "v"(g) : "memory");
}
__device__ __forceinline__ void wait_async0() {
  asm volatile("s_wait_asynccnt 0x0" ::: "memory");
}
// Fence-free workgroup split barrier: no loadcnt/dscnt drain, unlike
// __syncthreads() which emits s_wait_loadcnt_dscnt 0x0 (kills pipelining).
__device__ __forceinline__ void wg_barrier() {
  asm volatile("s_barrier_signal -1\n\ts_barrier_wait -1" ::: "memory");
}
// Compiler-only ordering fence (per-wave LDS is in-order in hardware).
__device__ __forceinline__ void cfence() {
  asm volatile("" ::: "memory");
}
__device__ __forceinline__ unsigned lds_off(const void* p) {
  return (unsigned)(size_t)p;   // low 32 bits of generic addr == LDS offset
}

// A-operand (16x32, 16-bit): lane = 16*khalf + m.
__device__ __forceinline__ v16h load_tile_a(const _Float16* src, int ld) {
  const int lane = threadIdx.x & 31;
  const int m  = lane & 15;
  const int kh = lane >> 4;
  const _Float16* p = src + m * ld + kh * 8;
  V16H u;
  u.h[0] = *(const h8*)(p);
  u.h[1] = *(const h8*)(p + 16);
  return u.v;
}

// B-operand (32x16, 16-bit): lane = 16*khalf + n; column n contiguous in K.
__device__ __forceinline__ v16h load_tile_b(const _Float16* src, int ld) {
  const int lane = threadIdx.x & 31;
  const int n  = lane & 15;
  const int kh = lane >> 4;
  const _Float16* p = src + n * ld + kh * 8;
  V16H u;
  u.h[0] = *(const h8*)(p);
  u.h[1] = *(const h8*)(p + 16);
  return u.v;
}

// ---------------- conversion / transpose kernels ----------------
__global__ void cvt_f32_to_f16(const float* __restrict__ src,
                               _Float16* __restrict__ dst, int n) {
  int i = blockIdx.x * blockDim.x + threadIdx.x;
  if (i < n) dst[i] = (_Float16)src[i];
}

__global__ void transpose_cvt(const float* __restrict__ w,
                              _Float16* __restrict__ wT, int K, int N) {
  int i = blockIdx.x * blockDim.x + threadIdx.x;
  if (i < K * N) {
    int k = i / N, n = i - k * N;
    wT[n * K + k] = (_Float16)w[i];
  }
}

// ---------------- GEMM1: qkv = x @ w_qkv, scatter into q/k/vT ----------------
// K-loop unrolled x2 with ping-pong LDS A buffers and ping-pong B register
// sets (no cross-set copies -> partial loadcnt waits, real overlap).
__global__ __launch_bounds__(256)
void gemm_qkv(const _Float16* __restrict__ xh,
              const _Float16* __restrict__ wT,   // [3D][D]
              _Float16* __restrict__ qh,         // [B][H][S][HD]
              _Float16* __restrict__ kh2,        // [B][H][S][HD]
              _Float16* __restrict__ vT) {       // [B][H][HD][S]
  __shared__ __align__(16) _Float16 ldsA[2][16 * 32];
  const int tid  = threadIdx.x;
  const int wave = tid >> 5;
  const int lane = tid & 31;
  const int mbase = blockIdx.x * 16;
  const int nbase = (blockIdx.y * 8 + wave) * 64;
  const _Float16* arow = xh + mbase * DD;
  const _Float16* brow = wT + nbase * DD;

  auto stageA = [&](int sb, int kk) {
    if (tid < 64) {
      int r = tid >> 2, part = tid & 3;
      async_ld_b128(lds_off(&ldsA[sb][r * 32 + part * 8]),
                    arow + r * DD + kk + part * 8);
    }
  };

  stageA(0, 0);
  wait_async0();
  wg_barrier();

  v8f acc[4] = {};
  v16h b0[4], b1[4];
  #pragma unroll
  for (int t = 0; t < 4; ++t)
    b0[t] = load_tile_b(brow + t * 16 * DD, DD);

  for (int k0 = 0; k0 < DD; k0 += 64) {
    // ---- half 0: compute chunk k0 (ldsA[0], b0); prefetch k0+32 ----
    stageA(1, k0 + 32);
    #pragma unroll
    for (int t = 0; t < 4; ++t)
      b1[t] = load_tile_b(brow + t * 16 * DD + k0 + 32, DD);
    {
      v16h a = load_tile_a(&ldsA[0][0], 32);
      #pragma unroll
      for (int t = 0; t < 4; ++t)
        acc[t] = wmma_f16(a, b0[t], acc[t]);
    }
    wait_async0();
    wg_barrier();

    // ---- half 1: compute chunk k0+32 (ldsA[1], b1); prefetch k0+64 ----
    const bool more = (k0 + 64) < DD;
    if (more) {
      stageA(0, k0 + 64);
      #pragma unroll
      for (int t = 0; t < 4; ++t)
        b0[t] = load_tile_b(brow + t * 16 * DD + k0 + 64, DD);
    }
    {
      v16h a = load_tile_a(&ldsA[1][0], 32);
      #pragma unroll
      for (int t = 0; t < 4; ++t)
        acc[t] = wmma_f16(a, b1[t], acc[t]);
    }
    wait_async0();
    wg_barrier();
  }

  const int half = lane >> 4, n = lane & 15;
  #pragma unroll
  for (int t = 0; t < 4; ++t) {
    int ncol = nbase + t * 16 + n;
    int sec = ncol >> 10;            // 0=q,1=k,2=v
    int c = ncol & 1023;
    int h = c >> 6, d = c & 63;
    #pragma unroll
    for (int r = 0; r < 8; ++r) {
      int row = mbase + r + 8 * half;          // = b*S + s
      int b_ = row >> 11, s = row & (SS - 1);
      _Float16 val = (_Float16)acc[t][r];
      if (sec == 0)      qh [((b_ * HH + h) * SS + s) * HD + d] = val;
      else if (sec == 1) kh2[((b_ * HH + h) * SS + s) * HD + d] = val;
      else               vT [((b_ * HH + h) * HD + d) * SS + s] = val;
    }
  }
}

// ---------------- Flash attention ----------------
// Block = 4 waves = 4 adjacent 16-query tiles of the SAME (b,h); K/V chunks
// async-staged into double-buffered LDS, shared by all 4 waves.
__global__ __launch_bounds__(128)
void attn_flash(const _Float16* __restrict__ qh,
                const _Float16* __restrict__ kh2,
                const _Float16* __restrict__ vT,
                _Float16* __restrict__ attnh) {  // [B][S][D]
  __shared__ __align__(16) _Float16 kbuf[2][32 * 64];  // keys:  32 x HD
  __shared__ __align__(16) _Float16 vbuf[2][64 * 32];  // vT:   HD x 32
  __shared__ __align__(16) _Float16 pbuf[4][16 * 32];  // per-wave P tile
  const int tid  = threadIdx.x;
  const int wave = tid >> 5;
  const int lane = tid & 31;
  const int idx = blockIdx.x;                 // 0..1023
  const int qg = idx & 31;                    // S/64 groups
  const int h = (idx >> 5) & (HH - 1);
  const int b = idx >> 9;

  const _Float16* qb = qh  + ((b * HH + h) * SS + (qg * 4 + wave) * 16) * HD;
  const _Float16* kb = kh2 + (b * HH + h) * SS * HD;
  const _Float16* vb = vT  + (b * HH + h) * HD * SS;

  auto stage = [&](int sb, int j0) {
    const _Float16* kg = kb + j0 * HD;        // fully contiguous 4KB
    unsigned ko = lds_off(&kbuf[sb][0]);
    async_ld_b128(ko + tid * 16,         kg + tid * 8);
    async_ld_b128(ko + (tid + 128) * 16, kg + (tid + 128) * 8);
    unsigned vo = lds_off(&vbuf[sb][0]);
    int s1 = tid, s2 = tid + 128;             // slot = d*4 + part
    async_ld_b128(vo + s1 * 16, vb + (s1 >> 2) * SS + j0 + (s1 & 3) * 8);
    async_ld_b128(vo + s2 * 16, vb + (s2 >> 2) * SS + j0 + (s2 & 3) * 8);
  };

  // Q tile (16 x 64) as two A operands, loaded once from global.
  v16h aq0 = load_tile_a(qb, HD);
  v16h aq1 = load_tile_a(qb + 32, HD);

  float mst[8], lst[8];
  #pragma unroll
  for (int r = 0; r < 8; ++r) { mst[r] = -1e30f; lst[r] = 0.f; }
  v8f o[4] = {};

  const int half = lane >> 4, n = lane & 15;
  _Float16* pl = pbuf[wave];

  stage(0, 0);
  wait_async0();
  wg_barrier();

  for (int jc = 0; jc < SS / 32; ++jc) {
    const int buf = jc & 1;
    if (jc + 1 < SS / 32) stage(buf ^ 1, (jc + 1) * 32);  // overlap next chunk
    const _Float16* kl = &kbuf[buf][0];
    const _Float16* vl = &vbuf[buf][0];

    // ---- scores: two 16-key groups, HD=64 => 2 WMMAs each (B from LDS) ----
    v8f s0 = {}, s1 = {};
    s0 = wmma_f16(aq0, load_tile_b(kl, 64),           s0);
    s0 = wmma_f16(aq1, load_tile_b(kl + 32, 64),      s0);
    s1 = wmma_f16(aq0, load_tile_b(kl + 16 * 64, 64), s1);
    s1 = wmma_f16(aq1, load_tile_b(kl + 16 * 64 + 32, 64), s1);

    // ---- online softmax (rows in VGPR index r, cols in 16-lane halves) ----
    float p0[8], p1[8];
    #pragma unroll
    for (int r = 0; r < 8; ++r) {
      float v0 = s0[r] * 0.125f;           // 1/sqrt(64)
      float v1 = s1[r] * 0.125f;
      float rm = fmaxf(v0, v1);
      #pragma unroll
      for (int msk = 1; msk < 16; msk <<= 1)
        rm = fmaxf(rm, __shfl_xor(rm, msk, 32));
      float mn = fmaxf(mst[r], rm);
      float alpha = __expf(mst[r] - mn);
      float e0 = __expf(v0 - mn), e1 = __expf(v1 - mn);
      float rs = e0 + e1;
      #pragma unroll
      for (int msk = 1; msk < 16; msk <<= 1)
        rs += __shfl_xor(rs, msk, 32);
      lst[r] = lst[r] * alpha + rs;
      mst[r] = mn;
      p0[r] = e0; p1[r] = e1;
      #pragma unroll
      for (int t = 0; t < 4; ++t) o[t][r] = o[t][r] * alpha;
    }

    // ---- P: C-layout -> A-layout via per-wave LDS bounce ----
    // pbuf[wave] is private to this wave; per-wave LDS ops are in-order,
    // so only a compiler fence is needed (no workgroup barrier).
    #pragma unroll
    for (int r = 0; r < 8; ++r) {
      int row = r + 8 * half;
      pl[row * 32 + n]      = (_Float16)p0[r];
      pl[row * 32 + 16 + n] = (_Float16)p1[r];
    }
    cfence();
    v16h ap = load_tile_a(pl, 32);

    // ---- O += P(16x32) * V(32x64): 4 WMMAs (B from LDS) ----
    #pragma unroll
    for (int t = 0; t < 4; ++t)
      o[t] = wmma_f16(ap, load_tile_b(vl + (t * 16) * 32, 32), o[t]);

    wait_async0();        // staged chunk for next iteration is in LDS
    wg_barrier();
  }

  // ---- normalize rows and store f16 [B][S][D] ----
  #pragma unroll
  for (int r = 0; r < 8; ++r) {
    float inv = 1.0f / lst[r];
    int s = (qg * 4 + wave) * 16 + r + 8 * half;
    #pragma unroll
    for (int t = 0; t < 4; ++t) {
      int d = t * 16 + n;
      attnh[(b * SS + s) * DD + h * HD + d] = (_Float16)(o[t][r] * inv);
    }
  }
}

// ---------------- GEMM2: out = attn @ w_out + b_out (f32 out) ----------------
__global__ __launch_bounds__(256)
void gemm_out(const _Float16* __restrict__ attnh,
              const _Float16* __restrict__ woutT,  // [D][D] = W^T
              const float* __restrict__ bias,
              float* __restrict__ out) {
  __shared__ __align__(16) _Float16 ldsA[2][16 * 32];
  const int tid  = threadIdx.x;
  const int wave = tid >> 5;
  const int lane = tid & 31;
  const int mbase = blockIdx.x * 16;
  const int nbase = (blockIdx.y * 8 + wave) * 64;
  const _Float16* arow = attnh + mbase * DD;
  const _Float16* brow = woutT + nbase * DD;

  auto stageA = [&](int sb, int kk) {
    if (tid < 64) {
      int r = tid >> 2, part = tid & 3;
      async_ld_b128(lds_off(&ldsA[sb][r * 32 + part * 8]),
                    arow + r * DD + kk + part * 8);
    }
  };

  stageA(0, 0);
  wait_async0();
  wg_barrier();

  v8f acc[4] = {};
  v16h b0[4], b1[4];
  #pragma unroll
  for (int t = 0; t < 4; ++t)
    b0[t] = load_tile_b(brow + t * 16 * DD, DD);

  for (int k0 = 0; k0 < DD; k0 += 64) {
    stageA(1, k0 + 32);
    #pragma unroll
    for (int t = 0; t < 4; ++t)
      b1[t] = load_tile_b(brow + t * 16 * DD + k0 + 32, DD);
    {
      v16h a = load_tile_a(&ldsA[0][0], 32);
      #pragma unroll
      for (int t = 0; t < 4; ++t)
        acc[t] = wmma_f16(a, b0[t], acc[t]);
    }
    wait_async0();
    wg_barrier();

    const bool more = (k0 + 64) < DD;
    if (more) {
      stageA(0, k0 + 64);
      #pragma unroll
      for (int t = 0; t < 4; ++t)
        b0[t] = load_tile_b(brow + t * 16 * DD + k0 + 64, DD);
    }
    {
      v16h a = load_tile_a(&ldsA[1][0], 32);
      #pragma unroll
      for (int t = 0; t < 4; ++t)
        acc[t] = wmma_f16(a, b1[t], acc[t]);
    }
    wait_async0();
    wg_barrier();
  }

  const int half = lane >> 4, n = lane & 15;
  #pragma unroll
  for (int t = 0; t < 4; ++t) {
    int ncol = nbase + t * 16 + n;
    float bv = bias[ncol];
    #pragma unroll
    for (int r = 0; r < 8; ++r) {
      int row = mbase + r + 8 * half;
      out[row * DD + ncol] = acc[t][r] + bv;
    }
  }
}

extern "C" void kernel_launch(void* const* d_in, const int* in_sizes, int n_in,
                              void* d_out, int out_size, void* d_ws, size_t ws_size,
                              hipStream_t stream) {
  const float* x     = (const float*)d_in[0];
  const float* w_qkv = (const float*)d_in[1];
  const float* w_out = (const float*)d_in[2];
  const float* b_out = (const float*)d_in[3];
  float* out = (float*)d_out;

  char* ws = (char*)d_ws;
  _Float16* xh    = (_Float16*)ws; ws += (size_t)BS  * DD * 2;   // 8 MB
  _Float16* wqkvT = (_Float16*)ws; ws += (size_t)N3D * DD * 2;   // 6 MB
  _Float16* woutT = (_Float16*)ws; ws += (size_t)DD  * DD * 2;   // 2 MB
  _Float16* qh    = (_Float16*)ws; ws += (size_t)BS  * DD * 2;   // 8 MB
  _Float16* kh2   = (_Float16*)ws; ws += (size_t)BS  * DD * 2;   // 8 MB
  _Float16* vT    = (_Float16*)ws; ws += (size_t)BS  * DD * 2;   // 8 MB
  _Float16* attnh = (_Float16*)ws;                               // 8 MB

  {
    int n = BS * DD;
    cvt_f32_to_f16<<<(n + 255) / 256, 256, 0, stream>>>(x, xh, n);
  }
  {
    int n = DD * N3D;
    transpose_cvt<<<(n + 255) / 256, 256, 0, stream>>>(w_qkv, wqkvT, DD, N3D);
  }
  {
    int n = DD * DD;
    transpose_cvt<<<(n + 255) / 256, 256, 0, stream>>>(w_out, woutT, DD, DD);
  }

  gemm_qkv<<<dim3(BS / 16, N3D / 512), 256, 0, stream>>>(xh, wqkvT, qh, kh2, vT);
  attn_flash<<<BB * HH * (SS / 64), 128, 0, stream>>>(qh, kh2, vT, attnh);
  gemm_out<<<dim3(BS / 16, DD / 512), 256, 0, stream>>>(attnh, woutT, b_out, out);
}
